// PedTRTransformer_18494129176853
// MI455X (gfx1250) — matrix operand order repaired and checked by reading, not compiled, not run on hardware
//
#include <hip/hip_runtime.h>
#include <hip/hip_bf16.h>
#include <math.h>

typedef __attribute__((ext_vector_type(16))) _Float16 v16h;
typedef __attribute__((ext_vector_type(8)))  float    v8f;

#define ASTR 36   // A-tile LDS row stride in floats (144B, bank-spreading)
#define BSTR 36   // B-tile LDS row stride in halfs  (72B)

// =====================================================================
// Generic WMMA GEMM:
//   C[z][m][n] = act( alpha * sum_k A[z][m][k] * B'[k][n] + bias[n] + resid[z][m][n] )
//   B'[k][n] = bTrans ? B[n*ldb+k] : B[k*ldb+n]
// Block = 256 threads = 8 waves (4 M x 2 N); block tile 128x64; K-chunk 32.
// Each wave owns a 32x32 sub-tile: 2 A-frags x 2 B-frags -> 4
// v_wmma_f32_16x16x32_f16 per chunk with 2x register reuse of each frag.
// Interior tiles stage A via global_load_async_to_lds_b128 (ASYNCcnt,
// s_wait_asynccnt) keeping A as f32 in LDS (converted at frag load);
// B is converted f32->f16 on store. Edge tiles use guarded scalar loads.
// =====================================================================
__global__ __launch_bounds__(256)
void gemm_wmma_k(const float* __restrict__ A, long long lda, long long aBat,
                 const float* __restrict__ B, long long ldb, long long bBat, int bTrans,
                 const float* __restrict__ bias,
                 const float* __restrict__ resid, long long ldr, long long rBat,
                 float* __restrict__ C, long long ldc, long long cBat,
                 int M, int N, int K, float alpha, int act)
{
  __shared__ float    Asf[128 * ASTR];   // 18 KB
  __shared__ _Float16 Bs[64 * BSTR];     // 4.5 KB
  const int tid  = threadIdx.x;
  const int lane = tid & 31;
  const int wid  = tid >> 5;
  const int wm   = wid >> 1;      // 0..3 (32-row strip)
  const int wn   = wid & 1;       // 0..1 (32-col strip)
  const int hl   = lane >> 4;     // wave half
  const int l16  = lane & 15;
  const int bm   = blockIdx.y * 128;
  const int bn   = blockIdx.x * 64;
  const long long z = blockIdx.z;
  A += z * aBat;
  B += z * bBat;
  C += z * cBat;
  if (resid) resid += z * rBat;

  const bool edgeM = (bm + 128 > M);
  const bool edgeN = (bn + 64 > N);

  v8f acc[4] = {{}, {}, {}, {}};

  for (int k0 = 0; k0 < K; k0 += 32) {
    const bool edgeK = (k0 + 32 > K);
    __syncthreads();

    // ---------------- stage A tile: 128 rows x 32 k (f32 in LDS) ----------------
    if (!edgeM && !edgeK) {
      // 1024 float4 / 256 threads = 4 each; async global->LDS, 16B per lane
      #pragma unroll
      for (int i = 0; i < 4; ++i) {
        int e  = i * 256 + tid;
        int r  = e >> 3, k4 = (e & 7) * 4;
        const float* gp = &A[(long long)(bm + r) * lda + k0 + k4];
        unsigned ldsA = (unsigned)(unsigned long long)(const void*)&Asf[r * ASTR + k4];
        asm volatile("global_load_async_to_lds_b128 %0, %1, off"
                     :: "v"(ldsA), "v"(gp) : "memory");
      }
      asm volatile("s_wait_asynccnt 0x0" ::: "memory");
    } else {
      #pragma unroll
      for (int i = 0; i < 16; ++i) {
        int e = i * 256 + tid;
        int r = e >> 5, k = e & 31;
        int gr = bm + r, gk = k0 + k;
        Asf[r * ASTR + k] = (gr < M && gk < K) ? A[(long long)gr * lda + gk] : 0.f;
      }
    }

    // ---------------- stage B tile: 64 n x 32 k (f16, n-major in LDS) ----------------
    if (!edgeN && !edgeK) {
      if (bTrans) {
        int i0 = tid, i1 = tid + 256;
        int n0 = i0 >> 3, k0b = (i0 & 7) * 4;
        int n1 = i1 >> 3, k1b = (i1 & 7) * 4;
        float4 va = *(const float4*)&B[(long long)(bn + n0) * ldb + k0 + k0b];
        float4 vb = *(const float4*)&B[(long long)(bn + n1) * ldb + k0 + k1b];
        _Float16* pa = &Bs[n0 * BSTR + k0b];
        pa[0] = (_Float16)va.x; pa[1] = (_Float16)va.y;
        pa[2] = (_Float16)va.z; pa[3] = (_Float16)va.w;
        _Float16* pb = &Bs[n1 * BSTR + k1b];
        pb[0] = (_Float16)vb.x; pb[1] = (_Float16)vb.y;
        pb[2] = (_Float16)vb.z; pb[3] = (_Float16)vb.w;
      } else {
        // B is [K=32 rows][N]; float4 along n, scatter-transpose into LDS
        int i0 = tid, i1 = tid + 256;
        int rk0 = i0 >> 4, n0 = (i0 & 15) * 4;
        int rk1 = i1 >> 4, n1 = (i1 & 15) * 4;
        float4 va = *(const float4*)&B[(long long)(k0 + rk0) * ldb + bn + n0];
        float4 vb = *(const float4*)&B[(long long)(k0 + rk1) * ldb + bn + n1];
        Bs[(n0 + 0) * BSTR + rk0] = (_Float16)va.x;
        Bs[(n0 + 1) * BSTR + rk0] = (_Float16)va.y;
        Bs[(n0 + 2) * BSTR + rk0] = (_Float16)va.z;
        Bs[(n0 + 3) * BSTR + rk0] = (_Float16)va.w;
        Bs[(n1 + 0) * BSTR + rk1] = (_Float16)vb.x;
        Bs[(n1 + 1) * BSTR + rk1] = (_Float16)vb.y;
        Bs[(n1 + 2) * BSTR + rk1] = (_Float16)vb.z;
        Bs[(n1 + 3) * BSTR + rk1] = (_Float16)vb.w;
      }
    } else {
      #pragma unroll
      for (int i = 0; i < 8; ++i) {
        int e = i * 256 + tid;
        int n = e >> 5, k = e & 31;
        int gn = bn + n, gk = k0 + k;
        float v = 0.f;
        if (gn < N && gk < K)
          v = bTrans ? B[(long long)gn * ldb + gk] : B[(long long)gk * ldb + gn];
        Bs[n * BSTR + k] = (_Float16)v;
      }
    }

    // hint-prefetch next A chunk -> global_prefetch_b8
    if (k0 + 32 < K) {
      int r = tid >> 1, k = (tid & 1) * 16;   // rows 0..127
      if (bm + r < M)
        __builtin_prefetch(&A[(long long)(bm + r) * lda + (k0 + 32) + k], 0, 1);
    }
    __syncthreads();

    // A fragments (f32 LDS -> f16 regs): 16-bit A 16x32 VGPR layout
    v16h af[2], bf[2];
    #pragma unroll
    for (int ia = 0; ia < 2; ++ia) {
      const float* Ab = &Asf[(wm * 32 + ia * 16 + l16) * ASTR];
      #pragma unroll
      for (int i = 0; i < 8; ++i) {
        int kk = (i < 4 ? 2 * i : 16 + 2 * (i - 4)) + hl * 8;
        af[ia][2 * i]     = (_Float16)Ab[kk];
        af[ia][2 * i + 1] = (_Float16)Ab[kk + 1];
      }
    }
    // B fragments: cols wn*32 + jb*16
    #pragma unroll
    for (int jb = 0; jb < 2; ++jb) {
      const _Float16* Bb = &Bs[(wn * 32 + jb * 16 + l16) * BSTR];
      #pragma unroll
      for (int j = 0; j < 8; ++j) {
        int kk = (j < 4 ? 2 * j : 16 + 2 * (j - 4)) + hl * 8;
        bf[jb][2 * j]     = Bb[kk];
        bf[jb][2 * j + 1] = Bb[kk + 1];
      }
    }
    #pragma unroll
    for (int ia = 0; ia < 2; ++ia)
      #pragma unroll
      for (int jb = 0; jb < 2; ++jb)
        acc[ia * 2 + jb] = __builtin_amdgcn_wmma_f32_16x16x32_f16(
            false, af[ia], false, bf[jb], (short)0, acc[ia * 2 + jb], false, false);
  }

  // C/D layout: VGPR vv -> row vv (+8 for upper half), col = lane&15
  #pragma unroll
  for (int ia = 0; ia < 2; ++ia) {
    #pragma unroll
    for (int jb = 0; jb < 2; ++jb) {
      #pragma unroll
      for (int vv = 0; vv < 8; ++vv) {
        int gm = bm + wm * 32 + ia * 16 + vv + hl * 8;
        int gn = bn + wn * 32 + jb * 16 + l16;
        if (gm < M && gn < N) {
          float val = acc[ia * 2 + jb][vv] * alpha;
          if (bias)  val += bias[gn];
          if (resid) val += resid[(long long)gm * ldr + gn];
          if (act == 1)      val = fmaxf(val, 0.f);
          else if (act == 2) val = 1.f / (1.f + expf(-val));
          C[(long long)gm * ldc + gn] = val;
        }
      }
    }
  }
}

// =====================================================================
// LayerNorm over E=256 (one row per block, one feature per thread)
// =====================================================================
__global__ __launch_bounds__(256)
void layernorm_k(const float* __restrict__ x, const float* __restrict__ g,
                 const float* __restrict__ b, float* __restrict__ y, int relu)
{
  long long row = blockIdx.x;
  int tid = threadIdx.x;
  float v = x[row * 256 + tid];
  __shared__ float red[8];
  __shared__ float s_mu, s_var;
  float s = v;
  for (int o = 16; o; o >>= 1) s += __shfl_xor(s, o, 32);
  if ((tid & 31) == 0) red[tid >> 5] = s;
  __syncthreads();
  if (tid == 0) { float t = 0; for (int i = 0; i < 8; ++i) t += red[i]; s_mu = t * (1.f / 256.f); }
  __syncthreads();
  float d = v - s_mu;
  s = d * d;
  for (int o = 16; o; o >>= 1) s += __shfl_xor(s, o, 32);
  if ((tid & 31) == 0) red[tid >> 5] = s;
  __syncthreads();
  if (tid == 0) { float t = 0; for (int i = 0; i < 8; ++i) t += red[i]; s_var = t * (1.f / 256.f); }
  __syncthreads();
  float out = d * rsqrtf(s_var + 1e-5f) * g[tid] + b[tid];
  if (relu) out = fmaxf(out, 0.f);
  y[row * 256 + tid] = out;
}

// =====================================================================
// Row softmax (in place), row length L, one row per block
// =====================================================================
__global__ __launch_bounds__(256)
void softmax_k(float* __restrict__ x, int L)
{
  long long row = blockIdx.x;
  float* p = x + row * L;
  int tid = threadIdx.x;
  __shared__ float red[8];
  __shared__ float sval;
  float mx = -1e30f;
  for (int i = tid; i < L; i += 256) mx = fmaxf(mx, p[i]);
  for (int o = 16; o; o >>= 1) mx = fmaxf(mx, __shfl_xor(mx, o, 32));
  if ((tid & 31) == 0) red[tid >> 5] = mx;
  __syncthreads();
  if (tid == 0) { float m = red[0]; for (int i = 1; i < 8; ++i) m = fmaxf(m, red[i]); sval = m; }
  __syncthreads();
  float m = sval, s = 0.f;
  for (int i = tid; i < L; i += 256) { float e = expf(p[i] - m); p[i] = e; s += e; }
  for (int o = 16; o; o >>= 1) s += __shfl_xor(s, o, 32);
  if ((tid & 31) == 0) red[tid >> 5] = s;
  __syncthreads();
  if (tid == 0) { float t = 0; for (int i = 0; i < 8; ++i) t += red[i]; sval = t; }
  __syncthreads();
  float inv = 1.f / sval;
  for (int i = tid; i < L; i += 256) p[i] *= inv;
}

// =====================================================================
// Tiny 7-key camera attention (one thread per (batch, head, query-cam))
// qkv: (4500*7, 768) laid out [q|k|v], head h at dims h*32..h*32+31
// =====================================================================
__global__ __launch_bounds__(256)
void attn7_k(const float* __restrict__ qkv, const float* __restrict__ vm,
             float* __restrict__ out)
{
  int idx = blockIdx.x * 256 + threadIdx.x;
  if (idx >= 4500 * 56) return;
  int b = idx / 56, r = idx % 56, h = r / 7, qs = r % 7;
  const float* qp = qkv + ((long long)(b * 7 + qs)) * 768 + h * 32;
  float sc[7], mx = -1e30f;
  for (int s = 0; s < 7; ++s) {
    const float* kp = qkv + ((long long)(b * 7 + s)) * 768 + 256 + h * 32;
    float d = 0.f;
    #pragma unroll
    for (int i = 0; i < 32; ++i) d += qp[i] * kp[i];
    d *= 0.17677669529663689f;                  // 1/sqrt(32)
    d += (vm[b * 7 + s] > 0.5f) ? 0.f : -1e9f;  // key mask bias
    sc[s] = d; mx = fmaxf(mx, d);
  }
  float sum = 0.f;
  for (int s = 0; s < 7; ++s) { sc[s] = expf(sc[s] - mx); sum += sc[s]; }
  float inv = 1.f / sum;
  float o[32];
  #pragma unroll
  for (int i = 0; i < 32; ++i) o[i] = 0.f;
  for (int s = 0; s < 7; ++s) {
    const float* vp = qkv + ((long long)(b * 7 + s)) * 768 + 512 + h * 32;
    float pr = sc[s] * inv;
    #pragma unroll
    for (int i = 0; i < 32; ++i) o[i] += pr * vp[i];
  }
  float* op = out + ((long long)(b * 7 + qs)) * 256 + h * 32;
  #pragma unroll
  for (int i = 0; i < 32; ++i) op[i] = o[i];
}

// =====================================================================
// Projection + bilinear feature sampling; grid (q=900, cam=7, p=5), 256 ch
// =====================================================================
__global__ __launch_bounds__(256)
void sample_k(const float* __restrict__ img, const float* __restrict__ proj,
              const float* __restrict__ rpd, float* __restrict__ samp,
              float* __restrict__ vm)
{
  int q = blockIdx.x, cam = blockIdx.y, p = blockIdx.z, c = threadIdx.x;
  float rx = rpd[((long long)p * 900 + q) * 2 + 0] * 480.f;
  float ry = rpd[((long long)p * 900 + q) * 2 + 1] * 1440.f;
  const float* Pm = proj + cam * 9;
  float icx = Pm[0] * rx + Pm[1] * ry + Pm[2];
  float icy = Pm[3] * rx + Pm[4] * ry + Pm[5];
  float z   = Pm[6] * rx + Pm[7] * ry + Pm[8];
  float zi = fmaxf(z, 1e-5f);
  float gx = (icx / zi) / 1920.f; gx = (gx - 0.5f) * 2.f;
  float gy = (icy / zi) / 1080.f; gy = (gy - 0.5f) * 2.f;
  int valid = (z > 1e-5f) && (gx > -1.f) && (gx < 1.f) && (gy > -1.f) && (gy < 1.f);
  if (c == 0) vm[((long long)p * 900 + q) * 7 + cam] = valid ? 1.f : 0.f;
  float X = ((gx + 1.f) * 240.f - 1.f) * 0.5f;
  float Y = ((gy + 1.f) * 135.f - 1.f) * 0.5f;
  float x0 = floorf(X), y0 = floorf(Y), x1 = x0 + 1.f, y1 = y0 + 1.f;
  float wa = (x1 - X) * (y1 - Y), wb = (x1 - X) * (Y - y0);
  float wc = (X - x0) * (y1 - Y), wd = (X - x0) * (Y - y0);
  const float* base = img + ((long long)cam * 256 + c) * (135 * 240);
  auto tap = [&](float xf, float yf) -> float {
    bool v = (xf >= 0.f) && (xf < 240.f) && (yf >= 0.f) && (yf < 135.f);
    int xi = (int)xf, yi = (int)yf;
    if (xi < 0) xi = 0; if (xi > 239) xi = 239;
    if (yi < 0) yi = 0; if (yi > 134) yi = 134;
    float val = base[yi * 240 + xi];
    return v ? val : 0.f;
  };
  float out = wa * tap(x0, y0) + wb * tap(x0, y1) + wc * tap(x1, y0) + wd * tap(x1, y1);
  if (!__builtin_isfinite(out)) out = 0.f;   // nan_to_num
  samp[(((long long)p * 900 + q) * 7 + cam) * 256 + c] = out;
}

// =====================================================================
// rpd points + inverse-sigmoid cache
// =====================================================================
__device__ __forceinline__ float invsigf(float x)
{
  x = fminf(fmaxf(x, 0.f), 1.f);
  float x1 = fmaxf(x, 1e-5f), x2 = fmaxf(1.f - x, 1e-5f);
  return logf(x1 / x2);
}

__global__ void rpd_k(const float* __restrict__ ref, const float* __restrict__ offo,
                      float* __restrict__ rpd, float* __restrict__ iv2)
{
  int q = blockIdx.x * 256 + threadIdx.x;
  if (q >= 900) return;
  float rx = ref[q * 2], ry = ref[q * 2 + 1];
  float ivx = invsigf(rx), ivy = invsigf(ry);
  iv2[q * 2] = ivx; iv2[q * 2 + 1] = ivy;
  rpd[(0 * 900 + q) * 2 + 0] = rx;
  rpd[(0 * 900 + q) * 2 + 1] = ry;
  for (int p = 1; p < 5; ++p) {
    float ox = offo[q * 8 + (p - 1) * 2 + 0];
    float oy = offo[q * 8 + (p - 1) * 2 + 1];
    rpd[((long long)p * 900 + q) * 2 + 0] = 1.f / (1.f + expf(-(ivx + ox)));
    rpd[((long long)p * 900 + q) * 2 + 1] = 1.f / (1.f + expf(-(ivy + oy)));
  }
}

// =====================================================================
// Masked camera mean; amask combine; elementwise adds
// =====================================================================
__global__ __launch_bounds__(256)
void maskmean_k(const float* __restrict__ x, const float* __restrict__ vm,
                float* __restrict__ y)
{
  int bb = blockIdx.x, c = threadIdx.x;
  float s = 0.f, ms = 0.f;
  for (int k = 0; k < 7; ++k) {
    float m = vm[bb * 7 + k];
    s  += x[((long long)(bb * 7 + k)) * 256 + c] * m;
    ms += m;
  }
  y[(long long)bb * 256 + c] = s / ms;
}

__global__ __launch_bounds__(256)
void combine_k(const float* __restrict__ y, const float* __restrict__ am,
               float* __restrict__ o)
{
  int q = blockIdx.x, c = threadIdx.x;
  float s = 0.f;
  for (int p = 0; p < 5; ++p)
    s += am[q * 5 + p] * y[((long long)(p * 900 + q)) * 256 + c];
  o[(long long)q * 256 + c] = s;
}

__global__ void add2_k(const float* a, const float* b, float* o, int n)
{
  int i = blockIdx.x * 256 + threadIdx.x;
  if (i < n) o[i] = a[i] + b[i];
}

__global__ void add3_k(const float* a, const float* b, const float* c, float* o, int n)
{
  int i = blockIdx.x * 256 + threadIdx.x;
  if (i < n) o[i] = a[i] + b[i] + c[i];
}

// =====================================================================
// Host-side launch
// =====================================================================
typedef long long ll;

static inline void gemm(hipStream_t st,
    const float* A, ll lda, ll aBat,
    const float* B, ll ldb, ll bBat, int bT,
    const float* bias,
    const float* resid, ll ldr, ll rBat,
    float* C, ll ldc, ll cBat,
    int M, int N, int K, float alpha, int act, int Z)
{
  dim3 g((N + 63) / 64, (M + 127) / 128, Z);
  gemm_wmma_k<<<g, dim3(256), 0, st>>>(A, lda, aBat, B, ldb, bBat, bT,
                                       bias, resid, ldr, rBat,
                                       C, ldc, cBat, M, N, K, alpha, act);
}

extern "C" void kernel_launch(void* const* d_in, const int* in_sizes, int n_in,
                              void* d_out, int out_size, void* d_ws, size_t ws_size,
                              hipStream_t stream)
{
  (void)in_sizes; (void)out_size; (void)ws_size;
  const float* img   = (const float*)d_in[0];
  const float* proj  = (const float*)d_in[1];   // (1,7,3,3)
  const float* query = (const float*)d_in[2];
  const float* qpos  = (const float*)d_in[3];
  const float* refp  = (const float*)d_in[4];

  // params: jax tree_flatten order (dict keys sorted alphabetically at each level)
  const float* P_[80];
  for (int i = 0; i < 80; ++i) P_[i] = (i + 5 < n_in) ? (const float*)d_in[i + 5] : nullptr;
  const float *aq_bin = P_[0], *aq_bout = P_[1], *aq_win = P_[2], *aq_wout = P_[3];
  const float *fn_l1b = P_[4], *fn_l1w = P_[5], *fn_l2b = P_[6], *fn_l2w = P_[7];
  const float *fm_l1b = P_[8], *fm_l1w = P_[9], *fm_l2b = P_[10], *fm_l2w = P_[11];
  // img_layers[i] at base 12+i*12:
  //   ff.l1.b, ff.l1.w, ff.l2.b, ff.l2.w, ln1.b, ln1.g, ln2.b, ln2.g,
  //   mha.b_in, mha.b_out, mha.w_in, mha.w_out
  const float *ln1b = P_[60], *ln1g = P_[61];
  const float *ln2b = P_[62], *ln2g = P_[63];
  const float *ln3b = P_[64], *ln3g = P_[65];
  const float *of_l1b = P_[66], *of_l1w = P_[67], *of_l2b = P_[68], *of_l2w = P_[69];
  const float *op_b = P_[70], *op_w = P_[71];
  const float *pe_l1b = P_[72], *pe_l1w = P_[73], *pe_l2b = P_[74], *pe_l2w = P_[75];
  const float *pe_ln1b = P_[76], *pe_ln1g = P_[77], *pe_ln2b = P_[78], *pe_ln2g = P_[79];

  // deterministic bump allocator on d_ws
  float* wp = (float*)d_ws;
  auto alloc = [&](size_t n) { float* p = wp; wp += n; return p; };
  const int Qn = 900, E = 256, TOK = 31500, BAT = 4500;
  float* qsum  = alloc((size_t)Qn * E);
  float* qkvq  = alloc((size_t)Qn * 768);
  float* scr   = alloc((size_t)8 * Qn * Qn);
  float* attnq = alloc((size_t)Qn * E);
  float* q1    = alloc((size_t)Qn * E);
  float* qln   = alloc((size_t)Qn * E);
  float* offh  = alloc((size_t)Qn * E);
  float* offo  = alloc((size_t)Qn * 8);
  float* fmh   = alloc((size_t)Qn * E);
  float* amask = alloc((size_t)Qn * 5 + 64);
  float* rpd   = alloc((size_t)5 * Qn * 2);
  float* iv2   = alloc((size_t)Qn * 2 + 64);
  float* x     = alloc((size_t)TOK * E);
  float* vmask = alloc((size_t)BAT * 7 + 64);
  float* t     = alloc((size_t)TOK * E);
  float* h1    = alloc((size_t)TOK * E);
  float* qkvx  = alloc((size_t)TOK * 768);
  float* ao    = alloc((size_t)TOK * E);
  float* y     = alloc((size_t)BAT * E);
  float* comb  = alloc((size_t)Qn * E);
  float* outp  = alloc((size_t)Qn * E);
  float* peh   = alloc((size_t)Qn * E);
  float* pel   = alloc((size_t)Qn * E);
  float* pe2   = alloc((size_t)Qn * E);
  float* posf  = alloc((size_t)Qn * E);
  float* q2    = alloc((size_t)Qn * E);
  float* q3    = alloc((size_t)Qn * E);
  float* ffh   = alloc((size_t)Qn * E);
  float* q4    = alloc((size_t)Qn * E);

  const float isq32 = 0.17677669529663689f;

  // ---- query path ----
  add2_k<<<(Qn * E + 255) / 256, 256, 0, stream>>>(query, qpos, qsum, Qn * E);
  gemm(stream, qsum, E, 0, aq_win, E, 0, 1, aq_bin, nullptr, 0, 0,
       qkvq, 768, 0, Qn, 768, E, 1.f, 0, 1);
  gemm(stream, qkvq, 768, 32, qkvq + 256, 768, 32, 1, nullptr, nullptr, 0, 0,
       scr, Qn, (ll)Qn * Qn, Qn, Qn, 32, isq32, 0, 8);
  softmax_k<<<8 * Qn, 256, 0, stream>>>(scr, Qn);
  gemm(stream, scr, Qn, (ll)Qn * Qn, qkvq + 512, 768, 32, 0, nullptr, nullptr, 0, 0,
       attnq, E, 32, Qn, 32, Qn, 1.f, 0, 8);
  gemm(stream, attnq, E, 0, aq_wout, E, 0, 1, aq_bout, qsum, E, 0,
       q1, E, 0, Qn, E, E, 1.f, 0, 1);
  layernorm_k<<<Qn, 256, 0, stream>>>(q1, ln1g, ln1b, qln, 0);

  // offset / fuse_mask MLPs
  gemm(stream, qln, E, 0, of_l1w, E, 0, 1, of_l1b, nullptr, 0, 0,
       offh, E, 0, Qn, E, E, 1.f, 1, 1);
  gemm(stream, offh, E, 0, of_l2w, E, 0, 1, of_l2b, nullptr, 0, 0,
       offo, 8, 0, Qn, 8, E, 1.f, 0, 1);
  gemm(stream, qln, E, 0, fm_l1w, E, 0, 1, fm_l1b, nullptr, 0, 0,
       fmh, E, 0, Qn, E, E, 1.f, 1, 1);
  gemm(stream, fmh, E, 0, fm_l2w, E, 0, 1, fm_l2b, nullptr, 0, 0,
       amask, 5, 0, Qn, 5, E, 1.f, 2, 1);

  rpd_k<<<(Qn + 255) / 256, 256, 0, stream>>>(refp, offo, rpd, iv2);
  sample_k<<<dim3(Qn, 7, 5), 256, 0, stream>>>(img, proj, rpd, x, vmask);

  // ---- 4-layer camera transformer over (4500, 7, 256) ----
  for (int L = 0; L < 4; ++L) {
    const int b0 = 12 + L * 12;
    const float *ff1b = P_[b0 + 0], *ff1w = P_[b0 + 1];
    const float *ff2b = P_[b0 + 2], *ff2w = P_[b0 + 3];
    const float *l1b = P_[b0 + 4], *l1g = P_[b0 + 5];
    const float *l2b = P_[b0 + 6], *l2g = P_[b0 + 7];
    const float *mbin = P_[b0 + 8], *mbout = P_[b0 + 9];
    const float *mwin = P_[b0 + 10], *mwout = P_[b0 + 11];

    layernorm_k<<<TOK, 256, 0, stream>>>(x, l1g, l1b, t, 0);
    gemm(stream, t, E, 0, mwin, E, 0, 1, mbin, nullptr, 0, 0,
         qkvx, 768, 0, TOK, 768, E, 1.f, 0, 1);
    attn7_k<<<(BAT * 56 + 255) / 256, 256, 0, stream>>>(qkvx, vmask, ao);
    gemm(stream, ao, E, 0, mwout, E, 0, 1, mbout, x, E, 0,
         x, E, 0, TOK, E, E, 1.f, 0, 1);
    layernorm_k<<<TOK, 256, 0, stream>>>(x, l2g, l2b, t, 0);
    gemm(stream, t, E, 0, ff1w, E, 0, 1, ff1b, nullptr, 0, 0,
         h1, E, 0, TOK, E, E, 1.f, 1, 1);
    gemm(stream, h1, E, 0, ff2w, E, 0, 1, ff2b, x, E, 0,
         x, E, 0, TOK, E, E, 1.f, 0, 1);
  }

  // masked mean over cams, amask-weighted combine, out projection
  maskmean_k<<<BAT, 256, 0, stream>>>(x, vmask, y);
  combine_k<<<Qn, 256, 0, stream>>>(y, amask, comb);
  gemm(stream, comb, E, 0, op_w, E, 0, 1, op_b, nullptr, 0, 0,
       outp, E, 0, Qn, E, E, 1.f, 0, 1);

  // positional encoder: relu(LN(lin(relu(LN(lin(iv2))))))
  gemm(stream, iv2, 2, 0, pe_l1w, 2, 0, 1, pe_l1b, nullptr, 0, 0,
       peh, E, 0, Qn, E, 2, 1.f, 0, 1);
  layernorm_k<<<Qn, 256, 0, stream>>>(peh, pe_ln1g, pe_ln1b, pel, 1);
  gemm(stream, pel, E, 0, pe_l2w, E, 0, 1, pe_l2b, nullptr, 0, 0,
       pe2, E, 0, Qn, E, E, 1.f, 0, 1);
  layernorm_k<<<Qn, 256, 0, stream>>>(pe2, pe_ln2g, pe_ln2b, posf, 1);

  // final residual stack
  add3_k<<<(Qn * E + 255) / 256, 256, 0, stream>>>(outp, qln, posf, q2, Qn * E);
  layernorm_k<<<Qn, 256, 0, stream>>>(q2, ln2g, ln2b, q3, 0);
  gemm(stream, q3, E, 0, fn_l1w, E, 0, 1, fn_l1b, nullptr, 0, 0,
       ffh, E, 0, Qn, E, E, 1.f, 1, 1);
  gemm(stream, ffh, E, 0, fn_l2w, E, 0, 1, fn_l2b, q3, E, 0,
       q4, E, 0, Qn, E, E, 1.f, 0, 1);
  layernorm_k<<<Qn, 256, 0, stream>>>(q4, ln3g, ln3b, (float*)d_out, 0);
}
